// _ATTR_NETWORK_67224828117404
// MI455X (gfx1250) — matrix operand order repaired and checked by reading, not compiled, not run on hardware
//
#include <hip/hip_runtime.h>

// ---------------------------------------------------------------------------
// Fused "attribute cluster logits" for MI455X (gfx1250, wave32, WMMA).
//
//   logits[b,v] =  sum_s w[b,s] * softmax_s(scores)[b,s,v] * scores[b,s,v]
//                + dot(vec_emb[ids[b]], W[v])
//   scores[b,s,v] = dot(W[attr_ids[b,s]], W[v]),  w = attr_tf * (s < len)
//
// One kernel, two launches (item pass overwrites d_out, user pass adds).
// Compute-bound (~66 GFLOP total, W fits in L2) -> bf16 WMMA with f32 acc.
// ---------------------------------------------------------------------------

typedef __attribute__((ext_vector_type(16))) __bf16 v16bf;
typedef __attribute__((ext_vector_type(8)))  float  v8f;

#define B_   64
#define S_   100
#define V_   20000
#define E_   128
#define SPAD 112          // S padded to 7 WMMA M-tiles; row 100 = embedding vec
#define NT   128          // V-tile width (8 waves x 16 columns)
#define ASTR 136          // A LDS row stride in bf16 (68 dwords: conflict-free)
#define BSTR 132          // B LDS row stride in bf16

union Frag {
  unsigned long long q[4];  // 4 x ds_load_b64 -> 8 VGPRs of packed bf16 pairs
  v16bf v;
};

static __device__ __forceinline__ unsigned short f2bf(float f) {
  unsigned u = __float_as_uint(f);
  u += 0x7FFFu + ((u >> 16) & 1u);        // round-to-nearest-even
  return (unsigned short)(u >> 16);
}

static __device__ __forceinline__ void store4bf(unsigned short* p, float4 v) {
  unsigned lo = (unsigned)f2bf(v.x) | ((unsigned)f2bf(v.y) << 16);
  unsigned hi = (unsigned)f2bf(v.z) | ((unsigned)f2bf(v.w) << 16);
  *reinterpret_cast<unsigned long long*>(p) =
      (unsigned long long)lo | ((unsigned long long)hi << 32);
}

__global__ __launch_bounds__(256, 1)
void attr_cluster_kernel(const int*   __restrict__ attr_ids,   // [B,S]
                         const float* __restrict__ attr_tf,    // [B,S]
                         const int*   __restrict__ attr_lens,  // [B]
                         const int*   __restrict__ vec_ids,    // [B]
                         const float* __restrict__ W,          // [V,E]
                         const float* __restrict__ vec_emb,    // [*,E]
                         float*       __restrict__ out,        // [B,V]
                         int accumulate)
{
  __shared__ unsigned short As[SPAD * ASTR];  // gathered emb rows (bf16)
  __shared__ unsigned short Bs[NT * BSTR];    // W tile rows (bf16)
  __shared__ float wmask[SPAD];               // tf * seq-mask, 0 for pad rows

  const int tid = threadIdx.x;
  const int b   = blockIdx.y;
  const int v0  = blockIdx.x * NT;

  // --- masked tf weights -------------------------------------------------
  if (tid < SPAD) {
    float t = 0.f;
    if (tid < S_) {
      int len = attr_lens[b];
      if (tid < len) t = attr_tf[b * S_ + tid];
    }
    wmask[tid] = t;
  }
  // --- embedding vector as A-row 100 (GEMM computes its dot with W tile) --
  if (tid < E_) {
    int vid = vec_ids[b];
    As[100 * ASTR + tid] = f2bf(vec_emb[vid * E_ + tid]);
  }
  // --- zero pad rows 101..111 --------------------------------------------
  for (int i = tid; i < 11 * E_; i += 256)
    As[(101 + (i >> 7)) * ASTR + (i & 127)] = 0;

  // --- gather A rows: emb_b[s,:] = W[attr_ids[b,s],:]  (float4 loads) ----
  const float4* W4 = reinterpret_cast<const float4*>(W);
  for (int u = tid; u < S_ * (E_ / 4); u += 256) {
    int s  = u >> 5;                 // E/4 == 32
    int e4 = u & 31;
    int id = attr_ids[b * S_ + s];
    store4bf(&As[s * ASTR + e4 * 4], W4[id * (E_ / 4) + e4]);
  }
  // --- B tile: W rows v0..v0+127 (zero-fill past V) ----------------------
  for (int u = tid; u < NT * (E_ / 4); u += 256) {
    int n  = u >> 5;
    int e4 = u & 31;
    int v  = v0 + n;
    float4 val = make_float4(0.f, 0.f, 0.f, 0.f);
    if (v < V_) val = W4[v * (E_ / 4) + e4];
    store4bf(&Bs[n * BSTR + e4 * 4], val);
  }
  __syncthreads();

  const int lane = tid & 31;
  const int wave = tid >> 5;          // wave w owns output columns w*16..w*16+15
  const int half = lane >> 4;         // lane half selects K sub-range
  const int mrow = lane & 15;         // M (A) / N (B) index within tile
  const int nrow = wave * 16 + mrow;  // B-tile row (= output column)

  v8f acc[7] = {};                    // 7 M-tiles x 16x16 f32 accumulators

  // K pair offsets per fragment VGPR-pair, per the 16-bit A 16x32 layout:
  // vgpr{0,1}->K 0..3, {2,3}->4..7, {4,5}->16..19, {6,7}->20..23 (+8 hi half)
  const int kAoff0 = 0, kAoff1 = 4, kAoff2 = 16, kAoff3 = 20;

#pragma unroll
  for (int ks = 0; ks < 4; ++ks) {    // K = 128 in 4 steps of 32
    Frag bf;
#pragma unroll
    for (int p = 0; p < 4; ++p) {     // B 32x16: same K order as A (pairs 0..3 -> K 0,4,16,20; +8 hi half)
      int kb = (p & 1) * 4 + (p >> 1) * 16 + half * 8 + ks * 32;
      bf.q[p] = *reinterpret_cast<const unsigned long long*>(&Bs[nrow * BSTR + kb]);
    }
#pragma unroll
    for (int m = 0; m < 7; ++m) {
      Frag af;
      const unsigned short* arow = &As[(m * 16 + mrow) * ASTR + half * 8 + ks * 32];
      af.q[0] = *reinterpret_cast<const unsigned long long*>(arow + kAoff0);
      af.q[1] = *reinterpret_cast<const unsigned long long*>(arow + kAoff1);
      af.q[2] = *reinterpret_cast<const unsigned long long*>(arow + kAoff2);
      af.q[3] = *reinterpret_cast<const unsigned long long*>(arow + kAoff3);
      acc[m] = __builtin_amdgcn_wmma_f32_16x16x32_bf16(
          false, af.v, false, bf.v, (short)0, acc[m], false, false);
    }
  }

  // --- softmax over S (within-lane + one cross-half shuffle) -------------
  // C/D layout: vgpr j, lanes 0-15 -> M=j, lanes 16-31 -> M=8+j, N=lane%16.
  float mx = -3.402823466e38f;
#pragma unroll
  for (int m = 0; m < 7; ++m)
#pragma unroll
    for (int j = 0; j < 8; ++j) {
      int s = m * 16 + j + half * 8;
      if (s < S_) mx = fmaxf(mx, acc[m][j]);
    }
  mx = fmaxf(mx, __shfl_xor(mx, 16, 32));

  float se = 0.f, wsum = 0.f;
#pragma unroll
  for (int m = 0; m < 7; ++m)
#pragma unroll
    for (int j = 0; j < 8; ++j) {
      int s = m * 16 + j + half * 8;
      if (s < S_) {
        float sc = acc[m][j];
        float e  = __expf(sc - mx);
        se   += e;
        wsum += wmask[s] * e * sc;
      }
    }
  se   += __shfl_xor(se, 16, 32);
  wsum += __shfl_xor(wsum, 16, 32);

  float res = wsum / se;

  // embedding dot lives at s=100 -> mtile 6, vgpr 4, lanes 0-15
  if (half == 0) {
    int v = v0 + nrow;
    if (v < V_) {
      float r = res + acc[6][4];
      long idx = (long)b * V_ + v;
      if (accumulate) r += out[idx];
      out[idx] = r;
    }
  }
}

// ---------------------------------------------------------------------------

extern "C" void kernel_launch(void* const* d_in, const int* in_sizes, int n_in,
                              void* d_out, int out_size, void* d_ws, size_t ws_size,
                              hipStream_t stream) {
  (void)in_sizes; (void)n_in; (void)out_size; (void)d_ws; (void)ws_size;

  const int*   attr_item = (const int*)  d_in[0];
  const float* tf_item   = (const float*)d_in[1];
  const int*   lens_item = (const int*)  d_in[2];
  const int*   item_ids  = (const int*)  d_in[3];
  const int*   attr_user = (const int*)  d_in[4];
  const float* tf_user   = (const float*)d_in[5];
  const int*   lens_user = (const int*)  d_in[6];
  const int*   user_ids  = (const int*)  d_in[7];
  const float* W_item    = (const float*)d_in[8];
  const float* W_user    = (const float*)d_in[9];
  const float* user_emb  = (const float*)d_in[10];
  const float* item_emb  = (const float*)d_in[11];
  float* out = (float*)d_out;

  dim3 grid((V_ + NT - 1) / NT, B_);
  // item pass: writes (covers every (b,v)); pairs with item_emb @ W_item.T
  attr_cluster_kernel<<<grid, 256, 0, stream>>>(
      attr_item, tf_item, lens_item, item_ids, W_item, item_emb, out, 0);
  // user pass: accumulates; pairs with user_emb @ W_user.T
  attr_cluster_kernel<<<grid, 256, 0, stream>>>(
      attr_user, tf_user, lens_user, user_ids, W_user, user_emb, out, 1);
}
